// TFGroupViTAssignAttention_16655883174549
// MI455X (gfx1250) — matrix operand order repaired and checked by reading, not compiled, not run on hardware
//
#include <hip/hip_runtime.h>
#include <hip/hip_bf16.h>
#include <math.h>

// ---------------------------------------------------------------------------
// Shapes (fixed by the reference): B=32, G=64, S=4096, D=384.
// Heavy GEMMs: v_wmma_f32_16x16x32_bf16 (f32 accumulate).
// K staging in the logits kernel: global_load_async_to_lds_b128 + ASYNCcnt.
// ---------------------------------------------------------------------------

#define B_  32
#define G_  64
#define S_  4096
#define D_  384

typedef __attribute__((ext_vector_type(16))) __bf16 v16bf;
typedef __attribute__((ext_vector_type(8)))  __bf16 v8bf;
typedef __attribute__((ext_vector_type(8)))  float  v8f;

__device__ __forceinline__ __bf16 f2bf(float f) {
  unsigned int u = __builtin_bit_cast(unsigned int, f);
  u += 0x7FFFu + ((u >> 16) & 1u);           // round-to-nearest-even
  unsigned short h = (unsigned short)(u >> 16);
  return __builtin_bit_cast(__bf16, h);
}

__device__ __forceinline__ v8f wmma_bf16(v16bf a, v16bf b, v8f c) {
  return __builtin_amdgcn_wmma_f32_16x16x32_bf16(
      /*neg_a=*/false, a, /*neg_b=*/false, b,
      /*c_mod=*/(short)0, c, /*reuse_a=*/false, /*reuse_b=*/false);
}

// A fragment (16x32 bf16, contiguous-k layout, same k order as B): lane -> row m0+(lane&15);
// K values = 16 contiguous elements at d0 + 16*(lane>>4).
__device__ __forceinline__ v16bf load_a_bf16(const __bf16* A, int lda,
                                             int m0, int d0, int lane) {
  int m = m0 + (lane & 15);
  int h = (lane >> 4) << 4;                  // 0 or 16
  const __bf16* p = A + (size_t)m * lda + d0 + h;
  v8bf lo = *(const v8bf*)(p);
  v8bf hi = *(const v8bf*)(p + 8);
  v16bf r;
#pragma unroll
  for (int i = 0; i < 8; ++i) { r[i] = lo[i]; r[8 + i] = hi[i]; }
  return r;
}

// Same A layout, f32 source with per-row scale, converted to bf16.
__device__ __forceinline__ v16bf load_a_f32(const float* A, int lda,
                                            int m0, int d0, int lane,
                                            float scale) {
  int m = m0 + (lane & 15);
  int h = (lane >> 4) << 4;
  const float* p = A + (size_t)m * lda + d0 + h;
  v16bf r;
#pragma unroll
  for (int i = 0; i < 8; ++i) r[i]     = f2bf(p[i]      * scale);
#pragma unroll
  for (int i = 0; i < 8; ++i) r[8 + i] = f2bf(p[8 + i] * scale);
  return r;
}

// B fragment (32x16 bf16, K x N) from a column-major source:
// element (k, n) at Bs[n*ldb + k]; 16 contiguous bf16 (32 B) per lane.
__device__ __forceinline__ v16bf load_b_colmaj(const __bf16* Bs, int ldb,
                                               int n0, int d0, int lane) {
  int n  = n0 + (lane & 15);
  int kb = (lane >> 4) << 4;                 // 0 or 16
  return *(const v16bf*)(Bs + (size_t)n * ldb + d0 + kb);
}

// C layout: vgpr r holds (m = r + 8*(lane>>4), n = lane&15).

// ---------------------------------------------------------------------------
// Kernel 0: zero sum_v (f32) + count (int).
// ---------------------------------------------------------------------------
__global__ void zero_words(unsigned int* p, int n) {
  int i = blockIdx.x * blockDim.x + threadIdx.x;
  if (i < n) p[i] = 0u;
}

// ---------------------------------------------------------------------------
// Kernel 1: transpose + bf16-convert the four weight matrices.
// wt[w][n][k] = W_w[k][n]   (WMMA B-fragments become contiguous 32-B loads)
// ---------------------------------------------------------------------------
__global__ void prep_weights(const float* Wq, const float* Wk,
                             const float* Wv, const float* Wp, __bf16* wt) {
  int idx = blockIdx.x * 256 + threadIdx.x;  // 4*384*384 total
  int w   = idx / (D_ * D_);
  int rem = idx % (D_ * D_);
  int n = rem / D_, k = rem % D_;
  const float* W = (w == 0) ? Wq : (w == 1) ? Wk : (w == 2) ? Wv : Wp;
  wt[idx] = f2bf(W[k * D_ + n]);
}

// ---------------------------------------------------------------------------
// Kernel 2: q projection (M=2048 rows): Y = bf16(X @ W + b).
// One wave = 16x64 strip, A reused across 4 n-tiles. jobs=(M/16)*6.
// ---------------------------------------------------------------------------
__global__ __launch_bounds__(256)
void proj_gemm_bf16(const float* __restrict__ X, const __bf16* __restrict__ Wt,
                    const float* __restrict__ bias, __bf16* __restrict__ Y,
                    int M) {
  int lane = threadIdx.x & 31;
  int wave = threadIdx.x >> 5;
  int job  = blockIdx.x * 8 + wave;
  int mt = job / 6, ng = job % 6;
  if (mt * 16 >= M) return;                   // wave-uniform guard
  int m0 = mt * 16, n0 = ng * 64;

  v8f c[4] = {};
  for (int d0 = 0; d0 < D_; d0 += 32) {
    v16bf a = load_a_f32(X, D_, m0, d0, lane, 1.0f);
#pragma unroll
    for (int t = 0; t < 4; ++t) {
      v16bf b = load_b_colmaj(Wt, D_, n0 + t * 16, d0, lane);
      c[t] = wmma_bf16(a, b, c[t]);
    }
  }
  int nl = lane & 15;
  int mh = (lane >> 4) << 3;
#pragma unroll
  for (int t = 0; t < 4; ++t) {
    float bv = bias[n0 + t * 16 + nl];
#pragma unroll
    for (int r = 0; r < 8; ++r)
      Y[(size_t)(m0 + mh + r) * D_ + n0 + t * 16 + nl] = f2bf(c[t][r] + bv);
  }
}

// ---------------------------------------------------------------------------
// Kernel 3: FUSED k/v projection. One block = one 16-row key tile, staged
// once as bf16 in LDS (key is read from HBM exactly once). Waves 0-3 produce
// k = key@Wk+bk, waves 4-7 produce v = key@Wv+bv; 6 n-tiles per wave.
// A-fragments come from LDS (ds_load_b128).
// ---------------------------------------------------------------------------
__global__ __launch_bounds__(256)
void proj_kv_dual(const float* __restrict__ X,
                  const __bf16* __restrict__ Wtk, const __bf16* __restrict__ Wtv,
                  const float* __restrict__ bk, const float* __restrict__ bv,
                  __bf16* __restrict__ Yk, __bf16* __restrict__ Yv) {
  __shared__ __bf16 xs[16 * D_];              // 12 KB
  int tid  = threadIdx.x;
  int lane = tid & 31;
  int wave = tid >> 5;
  int m0   = blockIdx.x * 16;

  // Stage + convert the 16x384 f32 tile (coalesced; 24 elems/thread).
#pragma unroll
  for (int j = 0; j < 24; ++j) {
    int idx = tid + j * 256;                  // 0..6143
    xs[idx] = f2bf(X[(size_t)m0 * D_ + idx]);
  }
  __syncthreads();

  int base = wave * 6;                        // 48 tile-jobs: [mat=0..1][nt=0..23]
  int mat  = base / 24;                       // wave-uniform (0: k, 1: v)
  int ntb  = base % 24;
  const __bf16* Wt = mat ? Wtv : Wtk;

  v8f c[6] = {};
  for (int d0 = 0; d0 < D_; d0 += 32) {
    v16bf a = load_a_bf16(xs, D_, 0, d0, lane);   // from LDS
#pragma unroll
    for (int t = 0; t < 6; ++t) {
      v16bf b = load_b_colmaj(Wt, D_, (ntb + t) * 16, d0, lane);
      c[t] = wmma_bf16(a, b, c[t]);
    }
  }

  const float* bias = mat ? bv : bk;
  __bf16* Y = mat ? Yv : Yk;
  int nl = lane & 15;
  int mh = (lane >> 4) << 3;
#pragma unroll
  for (int t = 0; t < 6; ++t) {
    int n0 = (ntb + t) * 16;
    float bb = bias[n0 + nl];
#pragma unroll
    for (int r = 0; r < 8; ++r)
      Y[(size_t)(m0 + mh + r) * D_ + n0 + nl] = f2bf(c[t][r] + bb);
  }
}

// ---------------------------------------------------------------------------
// Kernel 4: fused logits + hard/soft softmax over G.
// Block = (batch, 256-token chunk). K-outer loop: each 32-wide K step stages
// a 256x32 bf16 K-chunk into LDS via global_load_async_to_lds_b128 (ASYNCcnt),
// then all 8 waves run WMMAs from LDS (each K element used by 4 g-tiles).
// Then per-token: argmax over G (hard assign + counts) and softmax over G.
// ---------------------------------------------------------------------------
__global__ __launch_bounds__(256)
void attn_assign(const __bf16* __restrict__ Qb, const __bf16* __restrict__ Kb,
                 float* __restrict__ soft_attn, int* __restrict__ assign_idx,
                 int* __restrict__ count) {
  __shared__ float  lg[G_ * 256];             // 64 KB logits
  __shared__ __bf16 kch[256 * 32];            // 16 KB K chunk (token-major)
  const float scale = 0.05103103630798287f;   // 1/sqrt(384)
  int b    = blockIdx.x;
  int s0   = blockIdx.y * 256;
  int tid  = threadIdx.x;
  int lane = tid & 31;
  int wave = tid >> 5;

  const __bf16* Q = Qb + (size_t)b * G_ * D_;
  const __bf16* K = Kb + (size_t)b * S_ * D_;

  // Wave tiles: gt = wave>>1 (uniform), st in [(wave&1)*8, +8).
  int gt  = wave >> 1;
  int stb = (wave & 1) * 8;
  v8f acc[8] = {};

  unsigned kbase = (unsigned)(unsigned long long)(uintptr_t)kch; // LDS byte addr

#pragma unroll 1
  for (int d0 = 0; d0 < D_; d0 += 32) {
    // Async-stage 256 tokens x 32 dims (16 KB): 4 x b128 per thread.
#pragma unroll
    for (int i = 0; i < 4; ++i) {
      int lb  = (tid + i * 256) * 16;         // byte offset in chunk
      int tok = lb >> 6;                      // 64 B per token row
      int rb  = lb & 63;
      unsigned lds = kbase + (unsigned)lb;
      unsigned long long ga =
          (unsigned long long)(uintptr_t)(K + (size_t)(s0 + tok) * D_) +
          (unsigned)(d0 * 2 + rb);
      asm volatile("global_load_async_to_lds_b128 %0, %1, off"
                   :: "v"(lds), "v"(ga) : "memory");
    }
    asm volatile("s_wait_asynccnt 0x0" ::: "memory");
    __syncthreads();

    v16bf a = load_a_bf16(Q, D_, gt * 16, d0, lane);
#pragma unroll
    for (int t = 0; t < 8; ++t) {
      int st = stb + t;
      v16bf bfrag =
          *(const v16bf*)(kch + (st * 16 + (lane & 15)) * 32 + ((lane >> 4) << 4));
      acc[t] = wmma_bf16(a, bfrag, acc[t]);
    }
    __syncthreads();                          // before chunk overwrite
  }

  // Spill scaled logits to LDS: (g = gt*16 + r + 8*(lane>>4), s = st*16 + n).
  {
    int nl = lane & 15;
    int mh = (lane >> 4) << 3;
#pragma unroll
    for (int t = 0; t < 8; ++t) {
      int st = stb + t;
#pragma unroll
      for (int r = 0; r < 8; ++r)
        lg[(gt * 16 + mh + r) * 256 + st * 16 + nl] = acc[t][r] * scale;
    }
  }
  __syncthreads();

  // Per-token reductions over G (thread = one token column; coalesced).
  int sl = tid;
  float mx = -INFINITY; int arg = 0;
#pragma unroll 1
  for (int g = 0; g < G_; ++g) {
    float v = lg[g * 256 + sl];
    if (v > mx) { mx = v; arg = g; }          // first-max tiebreak (argmax)
  }
  float sum = 0.f;
#pragma unroll 1
  for (int g = 0; g < G_; ++g) sum += __expf(lg[g * 256 + sl] - mx);
  float inv = 1.f / sum;
  size_t obase = ((size_t)b * G_) * S_ + (size_t)(s0 + sl);
#pragma unroll 1
  for (int g = 0; g < G_; ++g)
    soft_attn[obase + (size_t)g * S_] = __expf(lg[g * 256 + sl] - mx) * inv;

  assign_idx[b * S_ + s0 + sl] = arg;
  atomicAdd(&count[b * G_ + arg], 1);
}

// ---------------------------------------------------------------------------
// Kernel 5: segmented sum  sum_v[b,g,:] += v[b,s,:] for assigned tokens.
// LDS bins [64 x 192] f32 (48 KB), conflict-free (one d-column per thread),
// two d-halves, flushed with global_atomic_add_f32.
// ---------------------------------------------------------------------------
__global__ __launch_bounds__(256)
void scatter_sum(const __bf16* __restrict__ Vb, const int* __restrict__ assign_idx,
                 float* __restrict__ sum_v) {
  const int DC = 192;
  __shared__ float bins[G_ * DC];             // 48 KB
  __shared__ int gidx[256];
  int b  = blockIdx.x;
  int s0 = blockIdx.y * 256;
  int tid = threadIdx.x;
  gidx[tid] = assign_idx[b * S_ + s0 + tid];
  __syncthreads();

#pragma unroll 1
  for (int half = 0; half < 2; ++half) {
    int dbase = half * DC;
    for (int i = tid; i < G_ * DC; i += 256) bins[i] = 0.f;
    __syncthreads();
#pragma unroll 1
    for (int s = 0; s < 256; ++s) {           // serial tokens: no races
      int g = gidx[s];
      const __bf16* vrow = Vb + ((size_t)(b * S_ + s0 + s)) * D_ + dbase;
      if (tid < DC) bins[g * DC + tid] += (float)vrow[tid];
    }
    __syncthreads();
    for (int i = tid; i < G_ * DC; i += 256) {
      int g = i / DC, d = dbase + (i % DC);
      atomicAdd(&sum_v[((size_t)b * G_ + g) * D_ + d], bins[i]);
    }
    __syncthreads();
  }
}

// ---------------------------------------------------------------------------
// Kernel 6: out = (sum_v * 1/(count+1)) @ Wp + bp   (f32 output)
// Row scaling fused into the A-fragment load. ASSIGN_EPS = 1.0.
// ---------------------------------------------------------------------------
__global__ __launch_bounds__(256)
void final_proj(const float* __restrict__ sum_v, const int* __restrict__ count,
                const __bf16* __restrict__ Wpt, const float* __restrict__ bp,
                float* __restrict__ out) {
  int lane = threadIdx.x & 31;
  int wave = threadIdx.x >> 5;
  int job  = blockIdx.x * 8 + wave;
  int mt = job / 6, ng = job % 6;
  int m0 = mt * 16, n0 = ng * 64;             // m in [0, B*G)

  int m = m0 + (lane & 15);
  float rowInv = 1.0f / ((float)count[m] + 1.0f);

  v8f c[4] = {};
  for (int d0 = 0; d0 < D_; d0 += 32) {
    v16bf a = load_a_f32(sum_v, D_, m0, d0, lane, rowInv);
#pragma unroll
    for (int t = 0; t < 4; ++t) {
      v16bf b = load_b_colmaj(Wpt, D_, n0 + t * 16, d0, lane);
      c[t] = wmma_bf16(a, b, c[t]);
    }
  }
  int nl = lane & 15;
  int mh = (lane >> 4) << 3;
#pragma unroll
  for (int t = 0; t < 4; ++t) {
    float bv = bp[n0 + t * 16 + nl];
#pragma unroll
    for (int r = 0; r < 8; ++r)
      out[(size_t)(m0 + mh + r) * D_ + n0 + t * 16 + nl] = c[t][r] + bv;
  }
}

// ---------------------------------------------------------------------------
// Host-side launcher. Workspace layout (256-B aligned, ~198 MB):
//   wt (4x384x384 bf16) | q_bf16 | k_bf16 | v_bf16 | sum_v f32 | count | idx
// ---------------------------------------------------------------------------
extern "C" void kernel_launch(void* const* d_in, const int* in_sizes, int n_in,
                              void* d_out, int out_size, void* d_ws, size_t ws_size,
                              hipStream_t stream) {
  const float* query = (const float*)d_in[0];
  const float* key   = (const float*)d_in[1];
  const float* Wq = (const float*)d_in[2];
  const float* bq = (const float*)d_in[3];
  const float* Wk = (const float*)d_in[4];
  const float* bk = (const float*)d_in[5];
  const float* Wv = (const float*)d_in[6];
  const float* bv = (const float*)d_in[7];
  const float* Wp = (const float*)d_in[8];
  const float* bp = (const float*)d_in[9];

  float* out       = (float*)d_out;                 // [B*G*D] = 786432
  float* soft_attn = out + (size_t)B_ * G_ * D_;    // [B*G*S] = 8388608

  char* ws = (char*)d_ws;
  const size_t off_wt  = 0;                                  // 1,179,648 B
  const size_t off_q   = off_wt  + (size_t)4 * D_ * D_ * 2;  // 1,572,864 B
  const size_t off_k   = off_q   + (size_t)B_ * G_ * D_ * 2; // 100,663,296 B
  const size_t off_v   = off_k   + (size_t)B_ * S_ * D_ * 2; // 100,663,296 B
  const size_t off_sum = off_v   + (size_t)B_ * S_ * D_ * 2; // 3,145,728 B
  const size_t off_cnt = off_sum + (size_t)B_ * G_ * D_ * 4; // 8,192 B
  const size_t off_idx = off_cnt + (size_t)B_ * G_ * 4;      // 524,288 B

  __bf16* wt    = (__bf16*)(ws + off_wt);
  __bf16* wtq   = wt;
  __bf16* wtk   = wt + (size_t)1 * D_ * D_;
  __bf16* wtv   = wt + (size_t)2 * D_ * D_;
  __bf16* wtp   = wt + (size_t)3 * D_ * D_;
  __bf16* q_bf  = (__bf16*)(ws + off_q);
  __bf16* k_bf  = (__bf16*)(ws + off_k);
  __bf16* v_bf  = (__bf16*)(ws + off_v);
  float*  sum_v = (float*)(ws + off_sum);
  int*    count = (int*)(ws + off_cnt);
  int*    idx   = (int*)(ws + off_idx);

  // 0) zero sum_v + count (contiguous 32-bit words)
  {
    int nw = B_ * G_ * D_ + B_ * G_;
    zero_words<<<(nw + 255) / 256, 256, 0, stream>>>((unsigned int*)sum_v, nw);
  }
  // 1) transpose + bf16 weights
  prep_weights<<<(4 * D_ * D_) / 256, 256, 0, stream>>>(Wq, Wk, Wv, Wp, wt);
  // 2) q projection (jobs = (2048/16)*6 = 768, 8 waves/block)
  proj_gemm_bf16<<<((B_ * G_ / 16) * 6) / 8, 256, 0, stream>>>(query, wtq, bq, q_bf, B_ * G_);
  // 3) fused k/v projection: key read from HBM once (8192 blocks)
  proj_kv_dual<<<(B_ * S_) / 16, 256, 0, stream>>>(key, wtk, wtv, bk, bv, k_bf, v_bf);
  // 4) fused logits + hard/soft softmax over G (async K staging)
  attn_assign<<<dim3(B_, S_ / 256), 256, 0, stream>>>(q_bf, k_bf, soft_attn, idx, count);
  // 5) segmented token->group sum
  scatter_sum<<<dim3(B_, S_ / 256), 256, 0, stream>>>(v_bf, idx, sum_v);
  // 6) renormalize + final projection
  final_proj<<<((B_ * G_ / 16) * 6) / 8, 256, 0, stream>>>(sum_v, count, wtp, bp, out);
  (void)in_sizes; (void)n_in; (void)out_size; (void)ws_size;
}
